// M2R_25778393710941
// MI455X (gfx1250) — matrix operand order repaired and checked by reading, not compiled, not run on hardware
//
#include <hip/hip_runtime.h>

#define BATCH 4096
#define NVDIM 16384
#define NRDIM 10000
#define KDIM  128
#define PSZ   ((size_t)BATCH * KDIM)   // one partial proj buffer (floats)

typedef __attribute__((ext_vector_type(16))) __bf16 v16bf;
typedef __attribute__((ext_vector_type(8)))  float  v8f;

static __device__ inline void pack8(v16bf& v, int base, float4 lo, float4 hi) {
    v[base + 0] = (__bf16)lo.x; v[base + 1] = (__bf16)lo.y;
    v[base + 2] = (__bf16)lo.z; v[base + 3] = (__bf16)lo.w;
    v[base + 4] = (__bf16)hi.x; v[base + 5] = (__bf16)hi.y;
    v[base + 6] = (__bf16)hi.z; v[base + 7] = (__bf16)hi.w;
}

// Partial proj = M[:, half] @ W[:, half]^T accumulated in fp32 via bf16 WMMA.
// block = 256 threads (8 waves); wave w -> N-tile w (16 output cols = 16 W rows).
// blockIdx.x -> 64 M rows (four 16-row A-tiles per wave, B-frag reused 4x).
// blockIdx.y -> 0: Mi, 1: Mj.   blockIdx.z -> NV half (split-K partial).
__global__ __launch_bounds__(256) void proj_gemm_kernel(
    const float* __restrict__ Mi, const float* __restrict__ Mj,
    const float* __restrict__ W, float* __restrict__ ws)
{
    const float* __restrict__ M = blockIdx.y ? Mj : Mi;
    float* __restrict__ P = ws + (size_t)(blockIdx.y * 2 + blockIdx.z) * PSZ;

    const int wave = threadIdx.x >> 5;   // 0..7
    const int lane = threadIdx.x & 31;
    const int l16  = lane & 15;
    const int lhi  = lane >> 4;          // 0 or 1

    const int rowBase = blockIdx.x * 64;
    const int nBase   = wave * 16;
    const int kStart  = blockIdx.z * (NVDIM / 2);
    const int kEnd    = kStart + (NVDIM / 2);

    const float* __restrict__ ap[4];
#pragma unroll
    for (int t = 0; t < 4; ++t)
        ap[t] = M + (size_t)(rowBase + t * 16 + l16) * NVDIM;
    const float* __restrict__ bp = W + (size_t)(nBase + l16) * NVDIM;

    v8f c[4] = {v8f{}, v8f{}, v8f{}, v8f{}};

    const int kaOff = lhi * 8;    // A run1 lane offset
    const int kbOff = lhi * 8;    // B run lane offset

    // unroll 2: two independent fragment buffer sets let the scheduler fill
    // the 4 co-exec slots after each v_wmma (WAR hazard) with real cvt work
#pragma unroll 2
    for (int kk = kStart; kk < kEnd; kk += 32) {
        // keep the HBM stream ahead; WGP-scope so all 8 waves hit near cache
#pragma unroll
        for (int t = 0; t < 4; ++t)
            __builtin_prefetch(ap[t] + kk + 512, 0, 3);

        const int kb = kk + kbOff;
        float4 b0 = *(const float4*)(bp + kb);
        float4 b1 = *(const float4*)(bp + kb + 4);
        float4 b2 = *(const float4*)(bp + kb + 16);
        float4 b3 = *(const float4*)(bp + kb + 20);
        v16bf bfrag;
        pack8(bfrag, 0, b0, b1);
        pack8(bfrag, 8, b2, b3);

        const int ka1 = kk + kaOff;
        const int ka2 = kk + 16 + kaOff;
#pragma unroll
        for (int t = 0; t < 4; ++t) {
            float4 r1a = *(const float4*)(ap[t] + ka1);
            float4 r1b = *(const float4*)(ap[t] + ka1 + 4);
            float4 r2a = *(const float4*)(ap[t] + ka2);
            float4 r2b = *(const float4*)(ap[t] + ka2 + 4);
            v16bf afrag;
            pack8(afrag, 0, r1a, r1b);
            pack8(afrag, 8, r2a, r2b);
            c[t] = __builtin_amdgcn_wmma_f32_16x16x32_bf16(
                false, afrag, false, bfrag, (short)0, c[t], false, false);
        }
    }

    // C/D layout: element j of lane -> row (lhi?8:0)+j, col l16 (within tile)
    const int col = nBase + l16;
#pragma unroll
    for (int t = 0; t < 4; ++t) {
#pragma unroll
        for (int j = 0; j < 8; ++j) {
            int row = rowBase + t * 16 + lhi * 8 + j;
            P[(size_t)row * KDIM + col] = c[t][j];
        }
    }
}

// loss[b] = relu(dot(Pj[b], r[:,rp[b]]) - dot(Pi[b], r[:,ri[b]]) + 1); mean -> out
// Partial proj halves summed here (fp32-exact composition of the split-K GEMM).
__global__ __launch_bounds__(128) void loss_kernel(
    const float* __restrict__ ws,
    const int* __restrict__ ri, const int* __restrict__ rp,
    const float* __restrict__ r, float* __restrict__ out)
{
    __shared__ float red[128];
    const int b = blockIdx.x;
    const int k = threadIdx.x;
    const size_t idx = (size_t)b * KDIM + k;

    const float* Pi0 = ws;
    const float* Pi1 = ws + PSZ;
    const float* Pj0 = ws + 2 * PSZ;
    const float* Pj1 = ws + 3 * PSZ;

    const int irp = rp[b];
    const int iri = ri[b];
    float pj = Pj0[idx] + Pj1[idx];
    float pi = Pi0[idx] + Pi1[idx];
    float v = pj * r[(size_t)k * NRDIM + irp] - pi * r[(size_t)k * NRDIM + iri];

    red[k] = v;
    __syncthreads();
#pragma unroll
    for (int s = 64; s > 0; s >>= 1) {
        if (k < s) red[k] += red[k + s];
        __syncthreads();
    }
    if (k == 0) {
        float l = red[0] + 1.0f;
        l = l > 0.0f ? l : 0.0f;
        atomicAdd(out, l * (1.0f / (float)BATCH));
    }
}

extern "C" void kernel_launch(void* const* d_in, const int* in_sizes, int n_in,
                              void* d_out, int out_size, void* d_ws, size_t ws_size,
                              hipStream_t stream) {
    (void)in_sizes; (void)n_in; (void)out_size; (void)ws_size;
    const float* Mi = (const float*)d_in[0];
    const float* Mj = (const float*)d_in[1];
    const int*   ri = (const int*)d_in[2];
    const int*   rp = (const int*)d_in[3];
    const float* W  = (const float*)d_in[4];
    const float* r  = (const float*)d_in[5];
    float* out = (float*)d_out;
    float* ws  = (float*)d_ws;   // 4 partial proj buffers: Pi0, Pi1, Pj0, Pj1

    hipMemsetAsync(out, 0, sizeof(float), stream);

    dim3 g1(BATCH / 64, 2, 2);
    proj_gemm_kernel<<<g1, 256, 0, stream>>>(Mi, Mj, W, ws);

    loss_kernel<<<BATCH, 128, 0, stream>>>(ws, ri, rp, r, out);
}
